// GraphConvolutionSparse_1297080124151
// MI455X (gfx1250) — matrix operand order repaired and checked by reading, not compiled, not run on hardware
//
#include <hip/hip_runtime.h>
#include <hip/hip_bf16.h>

#define N_NODES 100000
#define IN_DIM  256
#define OUT_DIM 128
#define NNZ     1600000
#define KEEP_PROB 0.9f

typedef __bf16 v16bf __attribute__((ext_vector_type(16)));
typedef float  v8f   __attribute__((ext_vector_type(8)));

union BFrag { v16bf v; unsigned short u[16]; unsigned int d[8]; uint4 q[2]; };
union AccU  { v8f v; float f[8]; };

// ---------- helpers ----------
__device__ __forceinline__ unsigned short bf_bits(__bf16 x) {
    return __builtin_bit_cast(unsigned short, x);
}
__device__ __forceinline__ float bf2f(unsigned short h) {
    return __uint_as_float(((unsigned int)h) << 16);
}
__device__ __forceinline__ unsigned int rotl32(unsigned int x, int n) {
    return (x << n) | (x >> (32 - n));
}

// JAX threefry2x32-20 with key(42) => (k0,k1)=(0,42); uniform [0,1) per element.
__device__ float threefry_uniform(unsigned int i, unsigned int H) {
    const unsigned int which = (i >= H) ? 1u : 0u;
    const unsigned int pos   = which ? (i - H) : i;
    unsigned int x0 = pos;
    unsigned int x1 = pos + H;
    const unsigned int ks0 = 0u, ks1 = 42u;
    const unsigned int ks2 = 0x1BD11BDAu ^ ks0 ^ ks1;
    const int rA[4] = {13, 15, 26, 6};
    const int rB[4] = {17, 29, 16, 24};
    x0 += ks0; x1 += ks1;
#pragma unroll
    for (int j = 0; j < 4; ++j) { x0 += x1; x1 = rotl32(x1, rA[j]); x1 ^= x0; }
    x0 += ks1; x1 += ks2 + 1u;
#pragma unroll
    for (int j = 0; j < 4; ++j) { x0 += x1; x1 = rotl32(x1, rB[j]); x1 ^= x0; }
    x0 += ks2; x1 += ks0 + 2u;
#pragma unroll
    for (int j = 0; j < 4; ++j) { x0 += x1; x1 = rotl32(x1, rA[j]); x1 ^= x0; }
    x0 += ks0; x1 += ks1 + 3u;
#pragma unroll
    for (int j = 0; j < 4; ++j) { x0 += x1; x1 = rotl32(x1, rB[j]); x1 ^= x0; }
    x0 += ks1; x1 += ks2 + 4u;
#pragma unroll
    for (int j = 0; j < 4; ++j) { x0 += x1; x1 = rotl32(x1, rA[j]); x1 ^= x0; }
    x0 += ks2; x1 += ks0 + 5u;
    unsigned int bits = which ? x1 : x0;
    return __uint_as_float((bits >> 9) | 0x3f800000u) - 1.0f;
}

// ---------- kernels ----------
__global__ void zero_f32(float* __restrict__ p, size_t n) {
    size_t i = (size_t)blockIdx.x * blockDim.x + threadIdx.x;
    size_t stride = (size_t)gridDim.x * blockDim.x;
    for (; i < n; i += stride) p[i] = 0.0f;
}

// One-time W f32 -> bf16 in FRAGMENT-MAJOR layout:
// each lane's complete 8-dword B fragment is 32 contiguous bytes.
// For element (k,n):
//   ks = k>>5 (k-step), half = (k&31)>>4, r = (k&15)>>1, lo = k&1
//   t = n>>4 (n-tile), m = n&15, lane = half*16 + m, frag = ks*8 + t
//   ushort index = ((frag*32 + lane)*8 + r)*2 + lo
__global__ void convert_w(const float* __restrict__ W, unsigned short* __restrict__ Wbf) {
    int i = blockIdx.x * blockDim.x + threadIdx.x;
    if (i >= IN_DIM * OUT_DIM) return;
    int k = i >> 7;          // / OUT_DIM
    int n = i & 127;         // % OUT_DIM
    int ks   = k >> 5;
    int half = (k >> 4) & 1;
    int r    = (k & 15) >> 1;
    int lo   = k & 1;
    int t    = n >> 4;
    int m    = n & 15;
    int lane = half * 16 + m;
    int frag = ks * 8 + t;
    Wbf[(((frag * 32 + lane) * 8) + r) * 2 + lo] = bf_bits((__bf16)W[i]);
}

// Sparse dropout + densify: Xd[row][col] += v  (1.6M atomics)
__global__ void scatter_features(const int* __restrict__ rows,
                                 const int* __restrict__ cols,
                                 const float* __restrict__ vals,
                                 float* __restrict__ Xd) {
    int i = blockIdx.x * blockDim.x + threadIdx.x;
    if (i >= NNZ) return;
    float u = threefry_uniform((unsigned int)i, NNZ / 2);
    bool keep = (KEEP_PROB + u) >= 1.0f;
    float v = keep ? vals[i] * (1.0f / KEEP_PROB) : 0.0f;
    if (v != 0.0f) {
        float* addr = Xd + (size_t)rows[i] * IN_DIM + cols[i];
        __hip_atomic_fetch_add(addr, v, __ATOMIC_RELAXED, __HIP_MEMORY_SCOPE_AGENT);
    }
}

// Dense GEMM h = Xd(100000x256,f32) @ W(256x128,bf16) -> Hbf (bf16), via WMMA bf16.
// Block: 256 threads = 8 waves; each wave computes a 16x128 strip.
__global__ __launch_bounds__(256) void gemm_xw(const float* __restrict__ Xd,
                                               const unsigned short* __restrict__ Wbf,
                                               unsigned short* __restrict__ Hbf) {
    __shared__ unsigned short sW[IN_DIM * OUT_DIM]; // 64 KB, fragment-major
    const int tid = threadIdx.x;
    // coalesced b128 copy: 64KB / 256 threads = 16 x uint4 per thread
    {
        const uint4* src = reinterpret_cast<const uint4*>(Wbf);
        uint4* dst = reinterpret_cast<uint4*>(sW);
#pragma unroll
        for (int i = 0; i < (IN_DIM * OUT_DIM / 8) / 256; ++i)
            dst[tid + i * 256] = src[tid + i * 256];
    }
    __syncthreads();

    const int wave = tid >> 5;
    const int lane = tid & 31;
    const int strip = blockIdx.x * 8 + wave;          // 16-row strip index
    const int n_strips = N_NODES / 16;                // 6250 exactly
    if (strip >= n_strips) return;                    // wave-uniform: EXEC stays all-ones

    const int m    = lane & 15;
    const int half = lane >> 4;
    const int row_base = strip * 16;
    const float* arow = Xd + (size_t)(row_base + m) * IN_DIM;
    const uint4* fragBase = reinterpret_cast<const uint4*>(sW) + lane * 2;

    AccU acc[8];
#pragma unroll
    for (int t = 0; t < 8; ++t)
#pragma unroll
        for (int j = 0; j < 8; ++j) acc[t].f[j] = 0.0f;

#pragma unroll
    for (int ks = 0; ks < 8; ++ks) {
        const int Kbase = ks * 32;
        // A fragment: one contiguous 16-float chunk (matches B packing)
        //   K in [Kbase+half*16, +16)
        const int kk0 = Kbase + half * 16;
        float4 q0 = *reinterpret_cast<const float4*>(arow + kk0);
        float4 q1 = *reinterpret_cast<const float4*>(arow + kk0 + 4);
        float4 q2 = *reinterpret_cast<const float4*>(arow + kk0 + 8);
        float4 q3 = *reinterpret_cast<const float4*>(arow + kk0 + 12);
        BFrag a;
        a.v[0]  = (__bf16)q0.x; a.v[1]  = (__bf16)q0.y;
        a.v[2]  = (__bf16)q0.z; a.v[3]  = (__bf16)q0.w;
        a.v[4]  = (__bf16)q1.x; a.v[5]  = (__bf16)q1.y;
        a.v[6]  = (__bf16)q1.z; a.v[7]  = (__bf16)q1.w;
        a.v[8]  = (__bf16)q2.x; a.v[9]  = (__bf16)q2.y;
        a.v[10] = (__bf16)q2.z; a.v[11] = (__bf16)q2.w;
        a.v[12] = (__bf16)q3.x; a.v[13] = (__bf16)q3.y;
        a.v[14] = (__bf16)q3.z; a.v[15] = (__bf16)q3.w;

#pragma unroll
        for (int t = 0; t < 8; ++t) {
            // fragment-major: 2x ds_load_b128 straight into 8 consecutive VGPRs
            const uint4* fp = fragBase + (size_t)(ks * 8 + t) * 64; // 32 lanes * 2 uint4
            BFrag b;
            b.q[0] = fp[0];
            b.q[1] = fp[1];
            acc[t].v = __builtin_amdgcn_wmma_f32_16x16x32_bf16(
                false, a.v, false, b.v, (short)0, acc[t].v, false, false);
        }
    }

    // Store D (C/D layout: VGPR j -> M = half*8 + j, N = t*16 + m) as bf16
#pragma unroll
    for (int t = 0; t < 8; ++t) {
        const int col = t * 16 + m;
#pragma unroll
        for (int j = 0; j < 8; ++j) {
            int row = row_base + half * 8 + j;
            Hbf[(size_t)row * OUT_DIM + col] = bf_bits((__bf16)acc[t].f[j]);
        }
    }
}

// SpMM#2: one wave per edge; gather bf16 h-row (L2 resident), 4 f32 atomics/lane.
__global__ __launch_bounds__(256) void spmm_adj(const int* __restrict__ arows,
                                                const int* __restrict__ acols,
                                                const float* __restrict__ avals,
                                                const unsigned short* __restrict__ Hbf,
                                                float* __restrict__ out) {
    const int gtid = blockIdx.x * blockDim.x + threadIdx.x;
    const int edge = gtid >> 5;
    const int lane = gtid & 31;
    if (edge >= NNZ) return;
    const int dst = arows[edge];
    const int src = acols[edge];
    const float w = avals[edge];

    // prefetch next edge's h row (global_prefetch_b8 path)
    if (edge + 1 < NNZ) {
        int nsrc = acols[edge + 1];
        __builtin_prefetch(Hbf + (size_t)nsrc * OUT_DIM + lane * 4, 0, 0);
    }

    // coalesced 256B row read: lane gets cols 4l..4l+3 (8 bytes)
    uint2 p = *reinterpret_cast<const uint2*>(Hbf + (size_t)src * OUT_DIM + lane * 4);
    float f0 = bf2f((unsigned short)(p.x & 0xffffu));
    float f1 = bf2f((unsigned short)(p.x >> 16));
    float f2 = bf2f((unsigned short)(p.y & 0xffffu));
    float f3 = bf2f((unsigned short)(p.y >> 16));

    float* orow = out + (size_t)dst * OUT_DIM + lane * 4;
    __hip_atomic_fetch_add(orow + 0, w * f0, __ATOMIC_RELAXED, __HIP_MEMORY_SCOPE_AGENT);
    __hip_atomic_fetch_add(orow + 1, w * f1, __ATOMIC_RELAXED, __HIP_MEMORY_SCOPE_AGENT);
    __hip_atomic_fetch_add(orow + 2, w * f2, __ATOMIC_RELAXED, __HIP_MEMORY_SCOPE_AGENT);
    __hip_atomic_fetch_add(orow + 3, w * f3, __ATOMIC_RELAXED, __HIP_MEMORY_SCOPE_AGENT);
}

__global__ void relu_inplace(float* __restrict__ p, size_t n) {
    size_t i = (size_t)blockIdx.x * blockDim.x + threadIdx.x;
    size_t stride = (size_t)gridDim.x * blockDim.x;
    for (; i < n; i += stride) p[i] = fmaxf(p[i], 0.0f);
}

// ---------- launch ----------
extern "C" void kernel_launch(void* const* d_in, const int* in_sizes, int n_in,
                              void* d_out, int out_size, void* d_ws, size_t ws_size,
                              hipStream_t stream) {
    const int*   feat_rows = (const int*)d_in[0];
    const int*   feat_cols = (const int*)d_in[1];
    const float* feat_vals = (const float*)d_in[2];
    const int*   adj_rows  = (const int*)d_in[3];
    const int*   adj_cols  = (const int*)d_in[4];
    const float* adj_vals  = (const float*)d_in[5];
    const float* W         = (const float*)d_in[6];
    float* out = (float*)d_out;

    // workspace layout: Xd f32 [N][256] | Hbf bf16 [N][128] | Wbf bf16 fragment-major
    char* ws = (char*)d_ws;
    float* Xd = (float*)ws;
    unsigned short* Hbf = (unsigned short*)(ws + (size_t)N_NODES * IN_DIM * sizeof(float));
    unsigned short* Wbf = (unsigned short*)(ws + (size_t)N_NODES * IN_DIM * sizeof(float)
                                               + (size_t)N_NODES * OUT_DIM * sizeof(unsigned short));

    const size_t xd_elems  = (size_t)N_NODES * IN_DIM;   // 25.6M
    const size_t out_elems = (size_t)N_NODES * OUT_DIM;  // 12.8M

    zero_f32<<<2048, 256, 0, stream>>>(Xd, xd_elems);
    zero_f32<<<2048, 256, 0, stream>>>(out, out_elems);

    convert_w<<<(IN_DIM * OUT_DIM + 255) / 256, 256, 0, stream>>>(W, Wbf);

    scatter_features<<<(NNZ + 255) / 256, 256, 0, stream>>>(feat_rows, feat_cols,
                                                            feat_vals, Xd);

    const int n_strips = N_NODES / 16;                   // 6250
    gemm_xw<<<(n_strips + 7) / 8, 256, 0, stream>>>(Xd, Wbf, Hbf);

    spmm_adj<<<NNZ / 8, 256, 0, stream>>>(adj_rows, adj_cols, adj_vals, Hbf, out);

    relu_inplace<<<2048, 256, 0, stream>>>(out, out_elems);
}
